// TernaryLinear_19533511262471
// MI455X (gfx1250) — matrix operand order repaired and checked by reading, not compiled, not run on hardware
//
#include <hip/hip_runtime.h>

// ---------------------------------------------------------------------------
// TernaryLinear on MI455X (gfx1250):
//   scale = mean(|W|) + 1e-8
//   Wq    = clip(round(W/scale), -1, 1)            (exact in bf16)
//   Y     = (X @ Wq^T) * scale                     (bf16 WMMA, f32 accum)
// M = 32768, N = 4096, K = 4096
// ---------------------------------------------------------------------------

#define IN_DIM  4096
#define OUT_DIM 4096
#define NROWS   32768      // BATCH * SEQ
#define EPS     1e-8f

#define BM 128             // block tile M
#define BN 256             // block tile N
#define BK 32              // K-step
#define LDS_STRIDE 40      // bf16 elems per LDS row (32 used + 8 pad => 80B, conflict-free)

typedef __attribute__((ext_vector_type(16))) __bf16        v16bf;
typedef __attribute__((ext_vector_type(2)))  __bf16        v2bf;
typedef __attribute__((ext_vector_type(2)))  float          v2f;
typedef __attribute__((ext_vector_type(8)))  float          v8f;
typedef __attribute__((ext_vector_type(16))) unsigned short v16u;
typedef __attribute__((ext_vector_type(8)))  unsigned short v8u;

// exact parameter type of the async-LDS builtins (per hipcc diagnostic):
// "int __attribute__((vector_size(16)))" in AS1 (global) / AS3 (LDS)
typedef int v4i __attribute__((vector_size(16)));
typedef v4i __attribute__((address_space(1)))* as1_v4i_p;
typedef v4i __attribute__((address_space(3)))* as3_v4i_p;

#if __has_builtin(__builtin_amdgcn_global_load_async_to_lds_b128)
#define HAVE_ASYNC_LDS 1
#else
#define HAVE_ASYNC_LDS 0
#endif

// two fp32 -> packed bf16x2 bits.
// Prefer the packed cvt builtin; else vector fptrunc so ISel can pick a
// packed v_cvt_pk_bf16_f32-class instruction (falls back to the integer
// RNE expansion only if the target truly lacks it).
static __device__ __forceinline__ unsigned pack_bf16(float a, float b) {
#if __has_builtin(__builtin_amdgcn_cvt_pk_bf16_f32)
    v2bf p = __builtin_amdgcn_cvt_pk_bf16_f32(a, b);
    return __builtin_bit_cast(unsigned, p);
#else
    v2f f;
    f[0] = a;
    f[1] = b;
    v2bf p = __builtin_convertvector(f, v2bf);
    return __builtin_bit_cast(unsigned, p);
#endif
}

static __device__ __forceinline__ void wait_async0() {
#if __has_builtin(__builtin_amdgcn_s_wait_asynccnt)
    __builtin_amdgcn_s_wait_asynccnt(0);
#else
    asm volatile("s_wait_asynccnt 0x0" ::: "memory");
#endif
}

// ---------------------------------------------------------------------------
// Stage 1a: per-block partial sums of |W| (fixed order => deterministic)
// ---------------------------------------------------------------------------
__global__ __launch_bounds__(256) void absmean_partial(const float* __restrict__ w,
                                                       float* __restrict__ partials) {
    __shared__ float red[256];
    size_t base = (size_t)blockIdx.x * 4096 + (size_t)threadIdx.x * 16;
    const float4* w4 = (const float4*)(w + base);
    float s = 0.0f;
#pragma unroll
    for (int i = 0; i < 4; ++i) {
        float4 v = w4[i];
        s += fabsf(v.x) + fabsf(v.y) + fabsf(v.z) + fabsf(v.w);
    }
    red[threadIdx.x] = s;
    __syncthreads();
#pragma unroll
    for (int off = 128; off > 0; off >>= 1) {
        if (threadIdx.x < (unsigned)off) red[threadIdx.x] += red[threadIdx.x + off];
        __syncthreads();
    }
    if (threadIdx.x == 0) partials[blockIdx.x] = red[0];
}

// ---------------------------------------------------------------------------
// Stage 1b: final reduction of 4096 partials -> scale
// ---------------------------------------------------------------------------
__global__ __launch_bounds__(256) void absmean_final(const float* __restrict__ partials,
                                                     float* __restrict__ scale_out) {
    __shared__ float red[256];
    float s = 0.0f;
#pragma unroll
    for (int i = 0; i < 16; ++i) s += partials[threadIdx.x + i * 256];
    red[threadIdx.x] = s;
    __syncthreads();
#pragma unroll
    for (int off = 128; off > 0; off >>= 1) {
        if (threadIdx.x < (unsigned)off) red[threadIdx.x] += red[threadIdx.x + off];
        __syncthreads();
    }
    if (threadIdx.x == 0)
        scale_out[0] = red[0] * (1.0f / 16777216.0f) + EPS;
}

// ---------------------------------------------------------------------------
// Stage 2: ternarize W -> bf16 bits {0xBF80, 0x0000, 0x3F80}
// ---------------------------------------------------------------------------
__global__ __launch_bounds__(256) void quantize_w(const float* __restrict__ w,
                                                  const float* __restrict__ scale_p,
                                                  uint2* __restrict__ wq) {
    const float inv = 1.0f / scale_p[0];
    size_t i = (size_t)blockIdx.x * 256 + threadIdx.x;   // one float4 each
    float4 v = ((const float4*)w)[i];
    uint2 o;
    o.x = pack_bf16(fminf(1.0f, fmaxf(-1.0f, rintf(v.x * inv))),
                    fminf(1.0f, fmaxf(-1.0f, rintf(v.y * inv))));
    o.y = pack_bf16(fminf(1.0f, fmaxf(-1.0f, rintf(v.z * inv))),
                    fminf(1.0f, fmaxf(-1.0f, rintf(v.w * inv))));
    wq[i] = o;
}

// ---------------------------------------------------------------------------
// Stage 3: GEMM  Y[M,N] = X[M,K] @ Wq[N,K]^T * scale
//   block: 128x256 tile, BK=32, 256 threads = 8 wave32 (2 x 4)
//   wave:  64x64 sub-tile = 4x4 v_wmma_f32_16x16x32_bf16 accumulators
//   B (weights) DMA'd global->LDS via async b128; A converted fp32->bf16 in VALU
//   double-buffered LDS, one barrier per K-step
// ---------------------------------------------------------------------------
__global__ __launch_bounds__(256) void tern_gemm(const float* __restrict__ X,
                                                 const unsigned short* __restrict__ Wq,
                                                 const float* __restrict__ scale_p,
                                                 float* __restrict__ Y) {
    __shared__ unsigned short Alds[2][BM * LDS_STRIDE];   // 2 x 10240 B
    __shared__ unsigned short Blds[2][BN * LDS_STRIDE];   // 2 x 20480 B

    const int t      = threadIdx.x;
    const int m_tile = blockIdx.y * BM;
    const int n_tile = blockIdx.x * BN;

    // --- A cooperative load: 2 threads per row, 16 fp32 each ---
    const int lrow = t >> 1;             // 0..127
    const int lcol = (t & 1) * 16;       // 0 or 16 (elements along K)
    // --- B: one 32-element (64 B) row per thread ---
    const int brow = t;                  // 0..255

    // --- wave / fragment indexing ---
    const int wid     = t >> 5;          // 0..7
    const int lane    = t & 31;
    const int wm      = wid & 1;         // wave row -> m offset = wm*64
    const int wn      = wid >> 1;        // wave col -> n offset = wn*64
    const int half_id = lane >> 4;       // selects contiguous K half (0..15 / 16..31)
    const int r16     = lane & 15;

    v8f acc[4][4] = {};

    const float*          Xg = X  + (size_t)(m_tile + lrow) * IN_DIM + lcol;
    const unsigned short* Wg = Wq + (size_t)(n_tile + brow) * IN_DIM;

    float4 ar[4];          // A prefetch registers (16 fp32)
#if !HAVE_ASYNC_LDS
    uint4  br[4];          // B prefetch registers (64 B) -- sync fallback only
#endif

    // issue B-tile copy for K-step kt into LDS buffer `buf`
    auto issue_b = [&](int buf, int kt) {
#if HAVE_ASYNC_LDS
        as1_v4i_p g1 = (as1_v4i_p)(Wg + kt * BK);
        as3_v4i_p l3 = (as3_v4i_p)&Blds[buf][brow * LDS_STRIDE];
        __builtin_amdgcn_global_load_async_to_lds_b128(g1, l3, 0,  0);
        __builtin_amdgcn_global_load_async_to_lds_b128(g1, l3, 16, 0);
        __builtin_amdgcn_global_load_async_to_lds_b128(g1, l3, 32, 0);
        __builtin_amdgcn_global_load_async_to_lds_b128(g1, l3, 48, 0);
#else
        const uint4* g = (const uint4*)(Wg + kt * BK);
#pragma unroll
        for (int i = 0; i < 4; ++i) br[i] = g[i];
#endif
    };

    auto load_a_regs = [&](int kt) {
        const float4* g = (const float4*)(Xg + kt * BK);
#pragma unroll
        for (int i = 0; i < 4; ++i) ar[i] = g[i];
    };

    // convert A regs to bf16 and commit tile data to LDS buffer `buf`
    auto commit = [&](int buf) {
        uint4 lo, hi;
        lo.x = pack_bf16(ar[0].x, ar[0].y);
        lo.y = pack_bf16(ar[0].z, ar[0].w);
        lo.z = pack_bf16(ar[1].x, ar[1].y);
        lo.w = pack_bf16(ar[1].z, ar[1].w);
        hi.x = pack_bf16(ar[2].x, ar[2].y);
        hi.y = pack_bf16(ar[2].z, ar[2].w);
        hi.z = pack_bf16(ar[3].x, ar[3].y);
        hi.w = pack_bf16(ar[3].z, ar[3].w);
        unsigned short* ap = &Alds[buf][lrow * LDS_STRIDE + lcol];
        *(uint4*)(ap)     = lo;
        *(uint4*)(ap + 8) = hi;
#if !HAVE_ASYNC_LDS
        unsigned short* bp = &Blds[buf][brow * LDS_STRIDE];
#pragma unroll
        for (int i = 0; i < 4; ++i) *(uint4*)(bp + i * 8) = br[i];
#endif
    };

    auto sync_buffers = [&]() {
#if HAVE_ASYNC_LDS
        wait_async0();
#endif
        __syncthreads();
    };

    // fragment loads + 16 WMMAs from LDS buffer `buf` (contiguous k order, k = 16*half + s)
    auto compute = [&](int buf) {
        v16bf a[4], b[4];
#pragma unroll
        for (int i = 0; i < 4; ++i) {
            // A 16x32: lane half 0 holds K 0..15, half 1 holds K 16..31
            const unsigned short* p =
                &Alds[buf][(wm * 64 + i * 16 + r16) * LDS_STRIDE + half_id * 16];
            v8u lo = *(const v8u*)p;
            v8u hi = *(const v8u*)(p + 8);
            v16u av;
#pragma unroll
            for (int e = 0; e < 8; ++e) { av[e] = lo[e]; av[e + 8] = hi[e]; }
            a[i] = __builtin_bit_cast(v16bf, av);
        }
#pragma unroll
        for (int j = 0; j < 4; ++j) {
            // B 32x16: column = r16, lane half selects K 0..15 vs 16..31
            const unsigned short* p =
                &Blds[buf][(wn * 64 + j * 16 + r16) * LDS_STRIDE + half_id * 16];
            b[j] = __builtin_bit_cast(v16bf, *(const v16u*)p);
        }
#pragma unroll
        for (int i = 0; i < 4; ++i)
#pragma unroll
            for (int j = 0; j < 4; ++j)
                acc[i][j] = __builtin_amdgcn_wmma_f32_16x16x32_bf16(
                    false, a[i], false, b[j], (short)0, acc[i][j], false, false);
    };

    // ---- prologue: fill buffer 0 ----
    issue_b(0, 0);
    load_a_regs(0);
    commit(0);
    sync_buffers();

    const int KT = IN_DIM / BK;
    for (int kt = 0; kt < KT; ++kt) {
        const int cur = kt & 1;
        const int nxt = cur ^ 1;
        const bool has_next = (kt + 1 < KT);
        if (has_next) {
            issue_b(nxt, kt + 1);     // async DMA into the other buffer
            load_a_regs(kt + 1);      // global fp32 loads in flight under compute
        }
        compute(cur);
        if (has_next) commit(nxt);    // convert + ds_store after WMMAs issued
        sync_buffers();
    }

    // ---- epilogue: scale and store (C layout: VGPR v -> M = half*8 + v, N = r16) ----
    const float s = scale_p[0];
#pragma unroll
    for (int i = 0; i < 4; ++i) {
        const int mb = m_tile + wm * 64 + i * 16 + half_id * 8;
#pragma unroll
        for (int j = 0; j < 4; ++j) {
            const int n = n_tile + wn * 64 + j * 16 + r16;
#pragma unroll
            for (int v = 0; v < 8; ++v)
                Y[(size_t)(mb + v) * OUT_DIM + n] = acc[i][j][v] * s;
        }
    }
}

// ---------------------------------------------------------------------------
// Launcher
// ---------------------------------------------------------------------------
extern "C" void kernel_launch(void* const* d_in, const int* in_sizes, int n_in,
                              void* d_out, int out_size, void* d_ws, size_t ws_size,
                              hipStream_t stream) {
    const float* x = (const float*)d_in[0];   // [8, 4096, 4096] fp32
    const float* w = (const float*)d_in[1];   // [4096, 4096] fp32
    float* out = (float*)d_out;               // [8, 4096, 4096] fp32

    char* ws = (char*)d_ws;
    float*          scale_p  = (float*)ws;                    // 4 B
    float*          partials = (float*)(ws + 256);            // 16 KB
    unsigned short* wq       = (unsigned short*)(ws + 32768); // 32 MB ternary bf16

    // 1) scale = mean(|W|) + eps   (two-stage, fixed-order, deterministic)
    absmean_partial<<<4096, 256, 0, stream>>>(w, partials);
    absmean_final<<<1, 256, 0, stream>>>(partials, scale_p);

    // 2) ternarize weights into bf16 bits
    quantize_w<<<16384, 256, 0, stream>>>(w, scale_p, (uint2*)wq);

    // 3) GEMM: x = N-tiles fastest so consecutive blocks share the same
    //    X tile through L2; Wq (32 MB) stays L2-resident.
    tern_gemm<<<dim3(OUT_DIM / BN, NROWS / BM), 256, 0, stream>>>(x, wq, scale_p, out);
}
